// TransformerMemoryUpdater_56599079027421
// MI455X (gfx1250) — compile-verified, matrix operand order checked by reading, not run on hardware
//
#include <hip/hip_runtime.h>
#include <hip/hip_bf16.h>

// ---------------- problem constants ----------------
#define NN     100000
#define MM     10
#define DIN    100
#define DT     100
#define DOUT   100
#define KD_KV  200          // DIN + DT
#define KP_KV  224          // padded to 7 * 32
#define KP_Q   128          // 100 padded to 4 * 32
#define NP     112          // 100 padded to 7 * 16
#define NB     16           // nodes per block (100000 % 16 == 0 -> 6250 blocks)
#define ROWS   (NB * MM)    // 160 mail rows per block
#define NT     7            // n tiles (112/16)
#define KT_KV  7            // k tiles for K/V gemm (224/32)
#define KT_Q   4            // k tiles for Q / MLP gemm (128/32)

typedef __attribute__((ext_vector_type(16))) __bf16 v16bf;
typedef __attribute__((ext_vector_type(8)))  __bf16 v8bf;
typedef __attribute__((ext_vector_type(8)))  float  v8f;

// ---------------- LDS layout (~152 KB -> 2 workgroups / 320KB WGP) ----------
struct __align__(32) SMem {
    __bf16 mails[ROWS][KP_KV];   // 160 x 224 bf16  = 71680 B (A operand for K/V)
    float  kv[ROWS][NP];         // 160 x 112 f32   = 71680 B (K result, then V)
    __bf16 nodeA[NB][KP_Q];      // 16 x 128 bf16   =  4096 B (mem, then normed rst)
    float  qbuf[NB][NP];         // 16 x 112 f32    =  7168 B (Q, then rst)
    float  att[ROWS][2];         //                 =  1280 B
    float  mu[NB], rs[NB];       //                 =   128 B
};

// ---------------- prep: swizzle fp32 weight into fragment-major bf16 panel --
// Panel element (kt, nt, lane, i) holds B[k][n] = W[n][k] with
//   k = kt*32 + (i>>3)*16 + (lane>>4)*8 + (i&7),  n = nt*16 + (lane&15)
// stored contiguously as (((kt*7 + nt)*32 + lane)*16 + i) so a wave's B
// fragment is a single 32B-per-lane coalesced load.
__global__ void build_panel(const float* __restrict__ W, __bf16* __restrict__ out,
                            int KD, int Kt) {
    int idx = blockIdx.x * blockDim.x + threadIdx.x;
    int total = Kt * NT * 32 * 16;
    if (idx >= total) return;
    int i    = idx & 15;
    int lane = (idx >> 4) & 31;
    int tile = idx >> 9;
    int nt   = tile % NT;
    int kt   = tile / NT;
    int k = kt * 32 + ((i >> 3) * 16) + ((lane >> 4) * 8) + (i & 7);
    int n = nt * 16 + (lane & 15);
    float v = (k < KD && n < DOUT) ? W[(size_t)n * KD + k] : 0.0f;
    out[idx] = (__bf16)v;
}

// ---------------- fragment loaders ----------------
// 16-bit A fragment (16x32), rows in LDS row-major with 'stride' bf16 elems.
// Per ISA: lane L holds row (L&15); VGPR0-3 = K 0..7 (lanes<16) / 8..15,
// VGPR4-7 = K 16..23 / 24..31  ->  two contiguous 16B LDS reads per lane.
__device__ __forceinline__ v16bf loadA(const __bf16* rowBase, int stride,
                                       int lane, int kBase) {
    const __bf16* p = rowBase + (size_t)(lane & 15) * stride + kBase + ((lane >> 4) * 8);
    v8bf lo = *(const v8bf*)(p);
    v8bf hi = *(const v8bf*)(p + 16);
    return __builtin_shufflevector(lo, hi, 0, 1, 2, 3, 4, 5, 6, 7,
                                           8, 9, 10, 11, 12, 13, 14, 15);
}

__device__ __forceinline__ v16bf loadB(const __bf16* panel, int kt, int nt, int lane) {
    return *(const v16bf*)(panel + (((size_t)(kt * NT + nt) * 32) + lane) * 16);
}

// ---------------- main fused kernel ----------------
__global__ __launch_bounds__(256)
void tgn_kernel(const float* __restrict__ mem, const float* __restrict__ mem_input,
                const float* __restrict__ ts, const float* __restrict__ mail_ts,
                const float* __restrict__ bq, const float* __restrict__ bk,
                const float* __restrict__ bv, const float* __restrict__ ln_g,
                const float* __restrict__ ln_b, const float* __restrict__ bm,
                const float* __restrict__ te_w, const float* __restrict__ te_b,
                const __bf16* __restrict__ Bq, const __bf16* __restrict__ Bk,
                const __bf16* __restrict__ Bv, const __bf16* __restrict__ Bm,
                float* __restrict__ out) {
    __shared__ SMem s;
    const int tid   = threadIdx.x;
    const int lane  = tid & 31;
    const int wave  = tid >> 5;
    const int node0 = blockIdx.x * NB;

    // hint next block's mail payload toward L2 (global_prefetch_b8)
    if (blockIdx.x + 1 < NN / NB)
        __builtin_prefetch(mem_input + (size_t)(node0 + NB) * (MM * DIN) + tid * 64, 0, 1);

    // ---- Phase A: stage mails (bf16) and mem rows (bf16) into LDS ----
    for (int idx = tid; idx < NB * MM * DIN; idx += 256) {
        int ln  = idx / (MM * DIN);
        int rem = idx - ln * (MM * DIN);
        int m   = rem / DIN, c = rem - m * DIN;
        float v = mem_input[(size_t)(node0 + ln) * (MM * DIN) + rem];
        s.mails[ln * MM + m][c] = (__bf16)v;
    }
    for (int idx = tid; idx < NB * MM * DT; idx += 256) {
        int ln  = idx / (MM * DT);
        int rem = idx - ln * (MM * DT);
        int m   = rem / DT, j = rem - m * DT;
        float dt = ts[node0 + ln] - mail_ts[(size_t)(node0 + ln) * MM + m];
        s.mails[ln * MM + m][DIN + j] = (__bf16)__cosf(dt * te_w[j] + te_b[j]);
    }
    for (int idx = tid; idx < ROWS * (KP_KV - KD_KV); idx += 256) {
        int r = idx / (KP_KV - KD_KV);
        s.mails[r][KD_KV + (idx % (KP_KV - KD_KV))] = (__bf16)0.0f;
    }
    for (int idx = tid; idx < NB * KP_Q; idx += 256) {
        int ln = idx / KP_Q, c = idx - ln * KP_Q;
        float v = (c < DOUT) ? mem[(size_t)(node0 + ln) * DOUT + c] : 0.0f;
        s.nodeA[ln][c] = (__bf16)v;
    }
    __syncthreads();

    // ---- Phase B: Q = mem @ Wq^T + bq  (1 row tile x 7 col tiles) ----
    if (wave < NT) {
        v8f acc = {};
#pragma unroll
        for (int kt = 0; kt < KT_Q; ++kt) {
            v16bf a = loadA(&s.nodeA[0][0], KP_Q, lane, kt * 32);
            v16bf b = loadB(Bq, kt, wave, lane);
            acc = __builtin_amdgcn_wmma_f32_16x16x32_bf16(false, a, false, b,
                                                          (short)0, acc, false, false);
        }
        int col  = wave * 16 + (lane & 15);
        float bb = (col < DOUT) ? bq[col] : 0.0f;
#pragma unroll
        for (int r = 0; r < 8; ++r)
            s.qbuf[r + ((lane >> 4) * 8)][col] = acc[r] + bb;
    }
    __syncthreads();

    // ---- Phase C: K = mails @ Wk^T + bk  (10 x 7 tiles over 8 waves) ----
    for (int job = wave; job < 10 * NT; job += 8) {
        int rt = job / NT, nt = job % NT;
        v8f acc = {};
#pragma unroll
        for (int kt = 0; kt < KT_KV; ++kt) {
            v16bf a = loadA(&s.mails[rt * 16][0], KP_KV, lane, kt * 32);
            v16bf b = loadB(Bk, kt, nt, lane);
            acc = __builtin_amdgcn_wmma_f32_16x16x32_bf16(false, a, false, b,
                                                          (short)0, acc, false, false);
        }
        int col  = nt * 16 + (lane & 15);
        float bb = (col < DOUT) ? bk[col] : 0.0f;
#pragma unroll
        for (int r = 0; r < 8; ++r)
            s.kv[rt * 16 + r + ((lane >> 4) * 8)][col] = acc[r] + bb;
    }
    __syncthreads();

    // ---- Phase D: attention logits (leaky-relu 0.2) + softmax over M ----
    for (int p = tid; p < ROWS * 2; p += 256) {
        int r = p >> 1, h = p & 1, ln = r / MM;
        float acc = 0.0f;
        for (int d = 0; d < DOUT / 2; ++d)
            acc += s.qbuf[ln][h * 50 + d] * s.kv[r][h * 50 + d];
        s.att[r][h] = (acc > 0.0f) ? acc : 0.2f * acc;
    }
    __syncthreads();
    if (tid < NB * 2) {
        int ln = tid >> 1, h = tid & 1;
        float mx = -1e30f;
        for (int m = 0; m < MM; ++m) mx = fmaxf(mx, s.att[ln * MM + m][h]);
        float e[MM], sum = 0.0f;
        for (int m = 0; m < MM; ++m) { e[m] = __expf(s.att[ln * MM + m][h] - mx); sum += e[m]; }
        float inv = 1.0f / sum;
        for (int m = 0; m < MM; ++m) s.att[ln * MM + m][h] = e[m] * inv;
    }
    __syncthreads();

    // ---- Phase E: V = mails @ Wv^T + bv  (overwrites s.kv) ----
    for (int job = wave; job < 10 * NT; job += 8) {
        int rt = job / NT, nt = job % NT;
        v8f acc = {};
#pragma unroll
        for (int kt = 0; kt < KT_KV; ++kt) {
            v16bf a = loadA(&s.mails[rt * 16][0], KP_KV, lane, kt * 32);
            v16bf b = loadB(Bv, kt, nt, lane);
            acc = __builtin_amdgcn_wmma_f32_16x16x32_bf16(false, a, false, b,
                                                          (short)0, acc, false, false);
        }
        int col  = nt * 16 + (lane & 15);
        float bb = (col < DOUT) ? bv[col] : 0.0f;
#pragma unroll
        for (int r = 0; r < 8; ++r)
            s.kv[rt * 16 + r + ((lane >> 4) * 8)][col] = acc[r] + bb;
    }
    __syncthreads();

    // ---- Phase F: rst = att-weighted V + mem  (into qbuf) ----
    for (int idx = tid; idx < NB * DOUT; idx += 256) {
        int ln = idx / DOUT, c = idx - ln * DOUT;
        int h  = c / 50;
        float acc = mem[(size_t)(node0 + ln) * DOUT + c];
        for (int m = 0; m < MM; ++m)
            acc += s.att[ln * MM + m][h] * s.kv[ln * MM + m][c];
        s.qbuf[ln][c] = acc;
    }
    __syncthreads();

    // ---- LayerNorm (eps=1e-5) -> bf16 into nodeA ----
    if (tid < NB) {
        float su = 0.0f, sq = 0.0f;
        for (int c = 0; c < DOUT; ++c) { float x = s.qbuf[tid][c]; su += x; sq += x * x; }
        float mu  = su / DOUT;
        float var = sq / DOUT - mu * mu;
        s.mu[tid] = mu;
        s.rs[tid] = rsqrtf(var + 1e-5f);
    }
    __syncthreads();
    for (int idx = tid; idx < NB * KP_Q; idx += 256) {
        int ln = idx / KP_Q, c = idx - ln * KP_Q;
        float v = 0.0f;
        if (c < DOUT)
            v = (s.qbuf[ln][c] - s.mu[ln]) * s.rs[ln] * ln_g[c] + ln_b[c];
        s.nodeA[ln][c] = (__bf16)v;
    }
    __syncthreads();

    // ---- Phase G: out = relu(rst @ Wm^T + bm), stored straight from D frags
    if (wave < NT) {
        v8f acc = {};
#pragma unroll
        for (int kt = 0; kt < KT_Q; ++kt) {
            v16bf a = loadA(&s.nodeA[0][0], KP_Q, lane, kt * 32);
            v16bf b = loadB(Bm, kt, wave, lane);
            acc = __builtin_amdgcn_wmma_f32_16x16x32_bf16(false, a, false, b,
                                                          (short)0, acc, false, false);
        }
        int col = wave * 16 + (lane & 15);
        if (col < DOUT) {
            float bb = bm[col];
#pragma unroll
            for (int r = 0; r < 8; ++r) {
                int ln  = r + ((lane >> 4) * 8);
                float v = acc[r] + bb;
                out[(size_t)(node0 + ln) * DOUT + col] = (v > 0.0f) ? v : 0.0f;
            }
        }
    }
}

// ---------------- launcher ----------------
extern "C" void kernel_launch(void* const* d_in, const int* in_sizes, int n_in,
                              void* d_out, int out_size, void* d_ws, size_t ws_size,
                              hipStream_t stream) {
    (void)in_sizes; (void)n_in; (void)out_size; (void)ws_size;
    const float* mem       = (const float*)d_in[0];
    const float* mem_input = (const float*)d_in[1];
    const float* ts        = (const float*)d_in[2];
    const float* mail_ts   = (const float*)d_in[3];
    const float* Wq        = (const float*)d_in[4];
    const float* bq        = (const float*)d_in[5];
    const float* Wk        = (const float*)d_in[6];
    const float* bk        = (const float*)d_in[7];
    const float* Wv        = (const float*)d_in[8];
    const float* bv        = (const float*)d_in[9];
    const float* ln_g      = (const float*)d_in[10];
    const float* ln_b      = (const float*)d_in[11];
    const float* Wm        = (const float*)d_in[12];
    const float* bm        = (const float*)d_in[13];
    const float* te_w      = (const float*)d_in[14];
    const float* te_b      = (const float*)d_in[15];

    __bf16* ws  = (__bf16*)d_ws;
    __bf16* Bk_ = ws;                 // 7*7*512 = 25088 bf16
    __bf16* Bv_ = ws + 25088;
    __bf16* Bq_ = ws + 50176;         // 4*7*512 = 14336 bf16
    __bf16* Bm_ = ws + 64512;         // total 78848 bf16 = 157696 B

    build_panel<<<(25088 + 255) / 256, 256, 0, stream>>>(Wk, Bk_, KD_KV, KT_KV);
    build_panel<<<(25088 + 255) / 256, 256, 0, stream>>>(Wv, Bv_, KD_KV, KT_KV);
    build_panel<<<(14336 + 255) / 256, 256, 0, stream>>>(Wq, Bq_, DOUT, KT_Q);
    build_panel<<<(14336 + 255) / 256, 256, 0, stream>>>(Wm, Bm_, DOUT, KT_Q);

    tgn_kernel<<<NN / NB, 256, 0, stream>>>(mem, mem_input, ts, mail_ts,
                                            bq, bk, bv, ln_g, ln_b, bm, te_w, te_b,
                                            Bq_, Bk_, Bv_, Bm_, (float*)d_out);
}
